// GroupAttention_81595788689467
// MI455X (gfx1250) — matrix operand order repaired and checked by reading, not compiled, not run on hardware
//
#include <hip/hip_runtime.h>
#include <hip/hip_bf16.h>

// GroupAttention for MI455X (gfx1250, wave32).
// B=4, S=8192, E=512, H=16, G=4, D=32, HPG=4. Tokens = B*S = 32768.
//
// Heavy work = 4 GEMMs [32768,512]x[512,512] -> run on v_wmma_f32_16x16x32_bf16.
// Per-token 4x4 group attention + argmax gating = tiny, done on VALU.

#define NTOK   32768
#define EDIM   512
#define NEGINF (-1e9f)

typedef __attribute__((ext_vector_type(16))) __bf16       v16bf;
typedef __attribute__((ext_vector_type(8)))  float        v8f;
typedef __attribute__((ext_vector_type(4)))  unsigned int v4u;

union FragU { v16bf v; v4u u[2]; };

__device__ __forceinline__ unsigned short f32_to_bf16(float f) {
    union { float f; unsigned int u; } c; c.f = f;
    unsigned int u = c.u + 0x7FFFu + ((c.u >> 16) & 1u);   // round-to-nearest-even
    return (unsigned short)(u >> 16);
}
__device__ __forceinline__ float bf16_to_f32(unsigned short h) {
    union { unsigned int u; float f; } c; c.u = ((unsigned int)h) << 16;
    return c.f;
}

// ---------------------------------------------------------------- conversions
__global__ __launch_bounds__(256) void cvt_bf16_kernel(
    const float* __restrict__ in, unsigned short* __restrict__ out, int n) {
    int i = blockIdx.x * 256 + threadIdx.x;
    if (i < n) out[i] = f32_to_bf16(in[i]);
}

// Wt[n*E + k] = bf16(W[k*E + n]) : weights pre-transposed to [N,K] so B
// fragments are contiguous 16B loads per lane.
__global__ __launch_bounds__(256) void transpose_w_kernel(
    const float* __restrict__ W, unsigned short* __restrict__ Wt) {
    int i = blockIdx.x * 256 + threadIdx.x;     // over E*E
    int n = i >> 9, k = i & 511;
    Wt[i] = f32_to_bf16(W[k * EDIM + n]);
}

// --------------------------------------------------------------- gating head
// One wave per token: gid = argmax_g (x . Wg[:,g] + bg[g]), first-max wins.
__global__ __launch_bounds__(256) void gid_kernel(
    const float* __restrict__ x, const float* __restrict__ Wg,
    const float* __restrict__ bg, int* __restrict__ gid) {
    int wave = (blockIdx.x * 256 + threadIdx.x) >> 5;
    int lane = threadIdx.x & 31;
    const float* row = x + (size_t)wave * EDIM;
    float a0 = 0.f, a1 = 0.f, a2 = 0.f, a3 = 0.f;
    for (int e = lane; e < EDIM; e += 32) {
        float xv = row[e];
        const float* w = Wg + e * 4;
        a0 += xv * w[0]; a1 += xv * w[1]; a2 += xv * w[2]; a3 += xv * w[3];
    }
    for (int off = 16; off >= 1; off >>= 1) {
        a0 += __shfl_xor(a0, off, 32);
        a1 += __shfl_xor(a1, off, 32);
        a2 += __shfl_xor(a2, off, 32);
        a3 += __shfl_xor(a3, off, 32);
    }
    if (lane == 0) {
        float v[4] = { a0 + bg[0], a1 + bg[1], a2 + bg[2], a3 + bg[3] };
        int best = 0; float bm = v[0];
        for (int i = 1; i < 4; i++) if (v[i] > bm) { bm = v[i]; best = i; }
        gid[wave] = best;
    }
}

// ----------------------------------------------------------------- WMMA GEMM
// A fragment (16x32 bf16, ISA §7.12.2): lane l holds row M=(l&15);
// lanes 0-15 carry K = kk+{0..7, 16..23}, lanes 16-31 K = kk+{8..15, 24..31}.
// Row-major A -> two contiguous 16B loads per lane.
__device__ __forceinline__ v16bf load_a_frag(
    const unsigned short* __restrict__ A, int mBase, int kk, int lane) {
    int row = mBase + (lane & 15);
    int k0  = kk + ((lane >> 4) << 3);
    const unsigned short* p = A + (size_t)row * EDIM + k0;
    FragU f;
    f.u[0] = *(const v4u*)(p);        // K = k0 .. k0+7
    f.u[1] = *(const v4u*)(p + 16);   // K = k0+16 .. k0+23
    return f.v;
}
// B fragment (32x16 bf16): lane l holds col N=(l&15); lanes 0-15 K=kk+{0..7, 16..23},
// lanes 16-31 K=kk+{8..15, 24..31} (2 bf16 per VGPR). Wt is [N,K] -> contiguous.
__device__ __forceinline__ v16bf load_b_frag(
    const unsigned short* __restrict__ Wt, int nBase, int kk, int lane) {
    int col = nBase + (lane & 15);
    int k0  = kk + ((lane >> 4) << 3);
    const unsigned short* p = Wt + (size_t)col * EDIM + k0;
    FragU f;
    f.u[0] = *(const v4u*)(p);
    f.u[1] = *(const v4u*)(p + 16);
    return f.v;
}

// Each wave computes a 32x32 C tile: 2x2 of 16x16, K-loop in steps of 32.
// Block = 256 threads = 8 waves -> 64(M) x 128(N) per block.
// Grid = (M/64, N/128) = (512, 4).
template <bool F32OUT>
__global__ __launch_bounds__(256) void gemm_bf16_kernel(
    const unsigned short* __restrict__ A,    // [M,512] bf16
    const unsigned short* __restrict__ Wt,   // [512,512] bf16, [N,K]
    const float* __restrict__ bias,          // [512] fp32
    void* __restrict__ out) {
    const int lane  = threadIdx.x & 31;
    const int wave  = threadIdx.x >> 5;
    const int mBase = blockIdx.x * 64  + (wave >> 2) * 32;
    const int nBase = blockIdx.y * 128 + (wave & 3)  * 32;

    v8f c00 = {}, c01 = {}, c10 = {}, c11 = {};
    for (int kk = 0; kk < EDIM; kk += 32) {
        if (kk + 32 < EDIM)   // global_prefetch_b8 for next A chunk
            __builtin_prefetch(A + (size_t)(mBase + (lane & 15)) * EDIM + kk + 32, 0, 1);
        v16bf a0 = load_a_frag(A, mBase,      kk, lane);
        v16bf a1 = load_a_frag(A, mBase + 16, kk, lane);
        v16bf b0 = load_b_frag(Wt, nBase,      kk, lane);
        v16bf b1 = load_b_frag(Wt, nBase + 16, kk, lane);
        c00 = __builtin_amdgcn_wmma_f32_16x16x32_bf16(false, a0, false, b0, (short)0, c00, false, false);
        c01 = __builtin_amdgcn_wmma_f32_16x16x32_bf16(false, a0, false, b1, (short)0, c01, false, false);
        c10 = __builtin_amdgcn_wmma_f32_16x16x32_bf16(false, a1, false, b0, (short)0, c10, false, false);
        c11 = __builtin_amdgcn_wmma_f32_16x16x32_bf16(false, a1, false, b1, (short)0, c11, false, false);
    }

    // C/D layout: element r of lane l -> row = (l>>4)*8 + r, col = l&15.
    const int rIn = (lane >> 4) * 8;
    const int col = lane & 15;
    v8f cc[2][2] = { { c00, c01 }, { c10, c11 } };
    for (int i = 0; i < 2; i++)
        for (int j = 0; j < 2; j++) {
            int colg = nBase + j * 16 + col;
            float bv = bias[colg];
            for (int r = 0; r < 8; r++) {
                int rowg  = mBase + i * 16 + rIn + r;
                float val = cc[i][j][r] + bv;
                if (F32OUT) ((float*)out)[(size_t)rowg * EDIM + colg] = val;
                else ((unsigned short*)out)[(size_t)rowg * EDIM + colg] = f32_to_bf16(val);
            }
        }
}

// ------------------------------------------------------- per-token attention
// One thread per (token, head): 4 scores over D=32, masked softmax, 4x32 AV.
// Masked-out groups get all scores = -1e9 -> uniform 0.25 weights (matches ref).
__global__ __launch_bounds__(256) void attn_kernel(
    const unsigned short* __restrict__ qb, const unsigned short* __restrict__ kb,
    const unsigned short* __restrict__ vb, const int* __restrict__ gid,
    unsigned short* __restrict__ attn) {
    int tid   = blockIdx.x * 256 + threadIdx.x;   // NTOK*16 threads
    int token = tid >> 4;
    int h     = tid & 15;
    int g     = h >> 2;
    const unsigned short* qrow = qb + (size_t)token * EDIM + h * 32;
    const unsigned short* krow = kb + (size_t)token * EDIM + g * 128;
    const unsigned short* vrow = vb + (size_t)token * EDIM + g * 128;

    float q[32];
    for (int d = 0; d < 32; d++) q[d] = bf16_to_f32(qrow[d]);

    const float scale = 0.1767766952966369f;      // 1/sqrt(32)
    const bool  ing   = (gid[token] == g);
    float s[4], m = -3.0e38f;
    for (int kh = 0; kh < 4; kh++) {
        float acc = 0.f;
        for (int d = 0; d < 32; d++) acc += q[d] * bf16_to_f32(krow[kh * 32 + d]);
        acc  = ing ? acc * scale : NEGINF;
        s[kh] = acc;
        m = fmaxf(m, acc);
    }
    float sum = 0.f;
    for (int kh = 0; kh < 4; kh++) { s[kh] = expf(s[kh] - m); sum += s[kh]; }
    float inv = 1.f / sum;
    for (int d = 0; d < 32; d++) {
        float acc = 0.f;
        for (int kh = 0; kh < 4; kh++) acc += s[kh] * bf16_to_f32(vrow[kh * 32 + d]);
        attn[(size_t)token * EDIM + h * 32 + d] = f32_to_bf16(acc * inv);
    }
}

// ------------------------------------------------------------------ launcher
extern "C" void kernel_launch(void* const* d_in, const int* in_sizes, int n_in,
                              void* d_out, int out_size, void* d_ws, size_t ws_size,
                              hipStream_t stream) {
    const float* x  = (const float*)d_in[0];
    const float* Wq = (const float*)d_in[1];
    const float* bq = (const float*)d_in[2];
    const float* Wk = (const float*)d_in[3];
    const float* bk = (const float*)d_in[4];
    const float* Wv = (const float*)d_in[5];
    const float* bv = (const float*)d_in[6];
    const float* Wg = (const float*)d_in[7];
    const float* bg = (const float*)d_in[8];
    const float* Wo = (const float*)d_in[9];
    const float* bo = (const float*)d_in[10];

    const size_t NE = (size_t)NTOK * EDIM;        // 16,777,216 elements

    // Workspace layout (~66 MB):
    //   [0, 32MB)   x_bf16  -> reused as attn output after QKV GEMMs
    //   [32, 64MB)  v_bf16
    //   [64MB..)    4x transposed bf16 weights (512KB each), then gid (128KB)
    char* ws = (char*)d_ws;
    unsigned short* xb   = (unsigned short*)ws;                   // x bf16 / attn
    unsigned short* vbuf = (unsigned short*)(ws + NE * 2);        // v
    unsigned short* WqT  = (unsigned short*)(ws + NE * 4);
    unsigned short* WkT  = WqT + (size_t)EDIM * EDIM;
    unsigned short* WvT  = WkT + (size_t)EDIM * EDIM;
    unsigned short* WoT  = WvT + (size_t)EDIM * EDIM;
    int*            gid  = (int*)(WoT + (size_t)EDIM * EDIM);

    // q and k are dead before the final GEMM writes d_out -> park them there.
    unsigned short* qbuf = (unsigned short*)d_out;                // 32 MB
    unsigned short* kbuf = qbuf + NE;                             // 32 MB

    // 1) conversions
    cvt_bf16_kernel<<<(int)(NE / 256), 256, 0, stream>>>(x, xb, (int)NE);
    transpose_w_kernel<<<(EDIM * EDIM) / 256, 256, 0, stream>>>(Wq, WqT);
    transpose_w_kernel<<<(EDIM * EDIM) / 256, 256, 0, stream>>>(Wk, WkT);
    transpose_w_kernel<<<(EDIM * EDIM) / 256, 256, 0, stream>>>(Wv, WvT);
    transpose_w_kernel<<<(EDIM * EDIM) / 256, 256, 0, stream>>>(Wo, WoT);

    // 2) gating head (fp32, wave per token)
    gid_kernel<<<NTOK * 32 / 256, 256, 0, stream>>>(x, Wg, bg, gid);

    // 3) Q/K/V projections on WMMA
    dim3 gg(NTOK / 64, EDIM / 128);
    gemm_bf16_kernel<false><<<gg, 256, 0, stream>>>(xb, WqT, bq, qbuf);
    gemm_bf16_kernel<false><<<gg, 256, 0, stream>>>(xb, WkT, bk, kbuf);
    gemm_bf16_kernel<false><<<gg, 256, 0, stream>>>(xb, WvT, bv, vbuf);

    // 4) per-token grouped attention (overwrites xb with attn output)
    attn_kernel<<<NTOK * 16 / 256, 256, 0, stream>>>(qbuf, kbuf, vbuf, gid, xb);

    // 5) output projection on WMMA, fp32 result into d_out
    gemm_bf16_kernel<true><<<gg, 256, 0, stream>>>(xb, WoT, bo, d_out);
}
